// BatchedAttentionAGaLiTeLayer_25933012533425
// MI455X (gfx1250) — matrix operand order repaired and checked by reading, not compiled, not run on hardware
//
#include <hip/hip_runtime.h>
#include <math.h>

// Problem constants (from reference)
#define TT 128
#define BB 32
#define IN_ 512
#define HH 8
#define DD 64
#define ETA_ 4
#define RR 4
#define DE_ 256          // ETA*D
#define MROWS 4096       // T*B
#define NKQV 2560        // H*D*5
#define NP 96            // H*ETA*3
#define KDIM 512

typedef __attribute__((ext_vector_type(16))) __bf16 v16bf;
typedef __attribute__((ext_vector_type(8)))  __bf16 v8bf;
typedef __attribute__((ext_vector_type(8)))  float  v8f;

__device__ __forceinline__ float sigm(float x) { return 1.0f / (1.0f + __expf(-x)); }

// ---------------------------------------------------------------------------
// f32 -> bf16 conversion
// ---------------------------------------------------------------------------
__global__ void cvt_f32_bf16(const float* __restrict__ in, __bf16* __restrict__ out, int n) {
  int i = blockIdx.x * blockDim.x + threadIdx.x;
  if (i < n) out[i] = (__bf16)in[i];
}

// ---------------------------------------------------------------------------
// WMMA bf16 GEMM with 2x2 register blocking:
//   Y[m,n] = sum_k A[m,k] * W[n,k] + bias[n]
// A: (M,K) row-major bf16, W: (N,K) row-major bf16, Y: (M,N) f32.
// One wave computes a 32x32 C tile (4 accumulators); 8 waves per block.
// Fragment layouts used here (contiguous k, not the ISA table order):
//  A (16x32): lane (hl,l16)=row; elem e -> k = hl*16+e  (matches B)
//  B (32x16): lane (hl,l16)=col; elem e -> k = hl*16+e  (one 32B run)
//  C/D: VGPR j, lane -> m = j + 8*hl, n = l16
// ---------------------------------------------------------------------------
__global__ __launch_bounds__(256) void gemm_bf16_wmma(
    const __bf16* __restrict__ A, const __bf16* __restrict__ W,
    const float* __restrict__ bias, float* __restrict__ Y,
    int M, int N, int K) {
  const int wid  = threadIdx.x >> 5;
  const int lane = threadIdx.x & 31;
  const int tile = blockIdx.x * 8 + wid;
  const int ntn  = N >> 5;                 // 32-wide tiles in N
  const int total = (M >> 5) * ntn;
  if (tile >= total) return;
  const int mt = tile / ntn;
  const int nt = tile % ntn;
  const int hl  = lane >> 4;               // half-wave select
  const int l16 = lane & 15;

  const __bf16* arow0 = A + (size_t)(mt * 32 + l16) * K;
  const __bf16* arow1 = A + (size_t)(mt * 32 + 16 + l16) * K;
  const __bf16* wrow0 = W + (size_t)(nt * 32 + l16) * K;
  const __bf16* wrow1 = W + (size_t)(nt * 32 + 16 + l16) * K;

  v8f c00 = {}, c01 = {}, c10 = {}, c11 = {};
  for (int kk = 0; kk < K; kk += 32) {
    __builtin_prefetch((const void*)(arow0 + kk + 128), 0, 1);
    __builtin_prefetch((const void*)(wrow0 + kk + 128), 0, 1);

    v8bf a00 = *(const v8bf*)(arow0 + kk + hl * 16);
    v8bf a01 = *(const v8bf*)(arow0 + kk + 8 + hl * 16);
    v8bf a10 = *(const v8bf*)(arow1 + kk + hl * 16);
    v8bf a11 = *(const v8bf*)(arow1 + kk + 8 + hl * 16);
    v16bf b0 = *(const v16bf*)(wrow0 + kk + hl * 16);
    v16bf b1 = *(const v16bf*)(wrow1 + kk + hl * 16);

    v16bf af0, af1;
#pragma unroll
    for (int e = 0; e < 8; ++e) {
      af0[e] = a00[e]; af0[e + 8] = a01[e];
      af1[e] = a10[e]; af1[e + 8] = a11[e];
    }
    c00 = __builtin_amdgcn_wmma_f32_16x16x32_bf16(false, af0, false, b0, (short)0, c00, false, false);
    c01 = __builtin_amdgcn_wmma_f32_16x16x32_bf16(false, af0, false, b1, (short)0, c01, false, false);
    c10 = __builtin_amdgcn_wmma_f32_16x16x32_bf16(false, af1, false, b0, (short)0, c10, false, false);
    c11 = __builtin_amdgcn_wmma_f32_16x16x32_bf16(false, af1, false, b1, (short)0, c11, false, false);
  }

  const int n0 = nt * 32 + l16;
  const int n1 = n0 + 16;
  const float bv0 = bias ? bias[n0] : 0.0f;
  const float bv1 = bias ? bias[n1] : 0.0f;
  float* y0 = Y + (size_t)(mt * 32 + 8 * hl) * N;        // rows m = base + j
  float* y1 = Y + (size_t)(mt * 32 + 16 + 8 * hl) * N;
#pragma unroll
  for (int j = 0; j < 8; ++j) {
    y0[(size_t)j * N + n0] = c00[j] + bv0;
    y0[(size_t)j * N + n1] = c01[j] + bv1;
    y1[(size_t)j * N + n0] = c10[j] + bv0;
    y1[(size_t)j * N + n1] = c11[j] + bv1;
  }
}

// ---------------------------------------------------------------------------
// Featurization: per (m = t*B+b, h) block, 256 threads over de = d*ETA + n.
// Produces keys_g, queries, dg (T,B,H,DE) and vb, db (T,B,H,D).
// ---------------------------------------------------------------------------
__global__ __launch_bounds__(256) void featurize(
    const float* __restrict__ kqv, const float* __restrict__ p,
    const float* __restrict__ term,
    float* __restrict__ kg, float* __restrict__ qy, float* __restrict__ dgo,
    float* __restrict__ vb, float* __restrict__ db) {
  const int mh = blockIdx.x;
  const int h = mh & (HH - 1);
  const int m = mh >> 3;
  const int de = threadIdx.x;
  const int d = de >> 2;       // outer
  const int n = de & 3;        // inner

  const float* kq = kqv + (size_t)m * NKQV + h * (5 * DD);
  const float kd = kq[d];
  const float qd = kq[64 + d];
  const float vd = kq[128 + d];
  const float bd = kq[192 + d];
  const float gd = kq[256 + d];

  const float* pp = p + (size_t)m * NP + h * 12;
  const float p1 = pp[n], p2 = pp[4 + n], p3 = pp[8 + n];

  const float nterm = 1.0f - term[m];   // term is (T,B) flat == m

  const float rk = fmaxf(kd, 0.0f) * fmaxf(p1, 0.0f);
  const float rq = fmaxf(qd, 0.0f) * fmaxf(p2, 0.0f);
  const float gm = sigm(gd) * sigm(p3);

  const size_t o = ((size_t)m * HH + h) * DE_ + de;
  kg[o]  = rk * gm;
  qy[o]  = rq;
  dgo[o] = (1.0f - gm) * nterm;

  if (n == 0) {
    const float bt = sigm(bd);
    const size_t ov = ((size_t)m * HH + h) * DD + d;
    vb[ov] = vd * bt;
    db[ov] = (1.0f - bt) * nterm;
  }
}

// ---------------------------------------------------------------------------
// Recurrence + attention readout. One block per (b,h), 256 threads = DE.
// Thread de owns K[r] (r=0..3) and S at index de, and V[rv][dv] with
// rv = de&3, dv = de>>2 so the R-reduction for kv is two in-wave shuffles.
// Time loop is software-pipelined (t+1 loads issued before t's reductions);
// wsum is double-buffered so only one barrier per step is needed.
// Writes attn (bf16, for output GEMM) and final states into d_out.
// ---------------------------------------------------------------------------
__global__ __launch_bounds__(256) void recurrence(
    const float* __restrict__ kg, const float* __restrict__ qy,
    const float* __restrict__ dgp, const float* __restrict__ vb,
    const float* __restrict__ db,
    const float* __restrict__ k0, const float* __restrict__ v0,
    const float* __restrict__ s0, const float* __restrict__ tick,
    __bf16* __restrict__ attnb, float* __restrict__ outbuf) {
  const int b = blockIdx.x >> 3;
  const int h = blockIdx.x & 7;
  const int de = threadIdx.x;
  const int lane = de & 31, wid = de >> 5;
  const int rv = de & 3, dv = de >> 2;

  float Ks[RR], S, V;
#pragma unroll
  for (int r = 0; r < RR; ++r)
    Ks[r] = k0[(((size_t)b * RR + r) * HH + h) * DE_ + de];
  V = v0[(((size_t)b * RR + rv) * HH + h) * DD + dv];
  S = s0[((size_t)b * HH + h) * DE_ + de];
  const float tk0 = tick[b];

  const float PI = 3.14159265358979323846f;
  const float omega[RR] = {-PI, -PI / 3.0f, PI / 3.0f, PI};

  __shared__ float wsum[2][5][8];

  // prologue loads for t = 0
  size_t base = (((size_t)b) * HH + h) * DE_ + de;
  size_t vbase = (((size_t)b) * HH + h) * DD + dv;
  const size_t tstride  = (size_t)BB * HH * DE_;
  const size_t tstridev = (size_t)BB * HH * DD;
  float q_c  = qy[base];
  float kx_c = kg[base];
  float dg_c = dgp[base];
  float vb_c = vb[vbase];
  float db_c = db[vbase];

  for (int t = 0; t < TT; ++t) {
    // issue next timestep's loads early (hide L2 latency behind reductions)
    const int tn = (t + 1 < TT) ? (t + 1) : t;
    const size_t nbase  = base + (size_t)tn * tstride - (size_t)t * 0;  // base is for t=0
    const size_t nb  = (((size_t)tn * BB + b) * HH + h) * DE_ + de;
    const size_t nvb = (((size_t)tn * BB + b) * HH + h) * DD + dv;
    float q_n  = qy[nb];
    float kx_n = kg[nb];
    float dg_n = dgp[nb];
    float vb_n = vb[nvb];
    float db_n = db[nvb];
    (void)nbase;

    const float ph = tk0 + (float)(t + 1);
    float occ[RR];
#pragma unroll
    for (int r = 0; r < RR; ++r) occ[r] = __cosf(ph * omega[r]);

    S = dg_c * S + kx_c;
    float part[5];
#pragma unroll
    for (int r = 0; r < RR; ++r) {
      Ks[r] = dg_c * Ks[r] + kx_c * occ[r];
      part[r] = Ks[r] * q_c;
    }
    part[4] = S * q_c;

    const int buf = t & 1;
#pragma unroll
    for (int j = 0; j < 5; ++j) {
      float v = part[j];
#pragma unroll
      for (int off = 16; off > 0; off >>= 1) v += __shfl_xor(v, off, 32);
      if (lane == 0) wsum[buf][j][wid] = v;
    }
    __syncthreads();

    float kdq[RR], norm;
#pragma unroll
    for (int j = 0; j < RR; ++j) {
      float s = 0.0f;
#pragma unroll
      for (int w = 0; w < 8; ++w) s += wsum[buf][j][w];
      kdq[j] = s;
    }
    {
      float s = 0.0f;
#pragma unroll
      for (int w = 0; w < 8; ++w) s += wsum[buf][4][w];
      norm = s;
    }

    // V update + kv reduction over r (threads de^1, de^2 share dv)
    V = db_c * V + vb_c * occ[rv];
    float kvs = V * kdq[rv];
    kvs += __shfl_xor(kvs, 1, 32);
    kvs += __shfl_xor(kvs, 2, 32);
    if (rv == 0) {
      const float a = kvs / (2.0f * RR * norm + 1e-5f);
      const size_t m = (size_t)t * BB + b;
      attnb[m * (HH * DD) + h * DD + dv] = (__bf16)a;
    }

    // rotate pipelined registers
    q_c = q_n; kx_c = kx_n; dg_c = dg_n; vb_c = vb_n; db_c = db_n;
  }

  // final states + tick into d_out (reference returns out, kf, vf, sf, tick+T)
  float* kf  = outbuf + 2097152;            // T*B*IN
  float* vf  = outbuf + 2097152 + 262144;   // + B*R*H*DE
  float* sf  = outbuf + 2097152 + 262144 + 65536;
  float* tko = outbuf + 2097152 + 262144 + 65536 + 65536;
#pragma unroll
  for (int r = 0; r < RR; ++r)
    kf[(((size_t)b * RR + r) * HH + h) * DE_ + de] = Ks[r];
  vf[(((size_t)b * RR + rv) * HH + h) * DD + dv] = V;
  sf[((size_t)b * HH + h) * DE_ + de] = S;
  if (h == 0 && de == 0) tko[b] = tk0 + (float)TT;
}

// ---------------------------------------------------------------------------
// Host launch
// ---------------------------------------------------------------------------
extern "C" void kernel_launch(void* const* d_in, const int* in_sizes, int n_in,
                              void* d_out, int out_size, void* d_ws, size_t ws_size,
                              hipStream_t stream) {
  const float* inputs = (const float*)d_in[0];
  const float* term   = (const float*)d_in[1];
  const float* k_prev = (const float*)d_in[2];
  const float* v_prev = (const float*)d_in[3];
  const float* s_prev = (const float*)d_in[4];
  const float* tick   = (const float*)d_in[5];
  const float* W_kqv  = (const float*)d_in[6];
  const float* b_kqv  = (const float*)d_in[7];
  const float* W_p    = (const float*)d_in[8];
  const float* b_p    = (const float*)d_in[9];
  const float* W_o    = (const float*)d_in[10];
  const float* b_o    = (const float*)d_in[11];
  float* out = (float*)d_out;

  // Workspace layout (all 256B aligned by construction)
  char* w = (char*)d_ws;
  size_t off = 0;
  __bf16* Xb    = (__bf16*)(w + off); off += (size_t)MROWS * KDIM * 2;        // 4 MiB
  __bf16* Wkqvb = (__bf16*)(w + off); off += (size_t)NKQV * KDIM * 2;         // 2.5 MiB
  __bf16* Wpb   = (__bf16*)(w + off); off += (size_t)NP * KDIM * 2;           // 96 KiB
  __bf16* Wob   = (__bf16*)(w + off); off += (size_t)IN_ * KDIM * 2;          // 512 KiB
  float*  kqv   = (float*)(w + off);  off += (size_t)MROWS * NKQV * 4;        // 40 MiB
  float*  p     = (float*)(w + off);  off += (size_t)MROWS * NP * 4;          // 1.5 MiB
  float*  kg    = (float*)(w + off);  off += (size_t)MROWS * HH * DE_ * 4;    // 32 MiB
  float*  qy    = (float*)(w + off);  off += (size_t)MROWS * HH * DE_ * 4;
  float*  dg    = (float*)(w + off);  off += (size_t)MROWS * HH * DE_ * 4;
  float*  vb    = (float*)(w + off);  off += (size_t)MROWS * HH * DD * 4;     // 8 MiB
  float*  db    = (float*)(w + off);  off += (size_t)MROWS * HH * DD * 4;
  __bf16* attnb = (__bf16*)(w + off); off += (size_t)MROWS * IN_ * 2;         // 4 MiB

  // 1) f32 -> bf16 conversions
  {
    int n = MROWS * KDIM;
    cvt_f32_bf16<<<(n + 255) / 256, 256, 0, stream>>>(inputs, Xb, n);
    n = NKQV * KDIM;
    cvt_f32_bf16<<<(n + 255) / 256, 256, 0, stream>>>(W_kqv, Wkqvb, n);
    n = NP * KDIM;
    cvt_f32_bf16<<<(n + 255) / 256, 256, 0, stream>>>(W_p, Wpb, n);
    n = IN_ * KDIM;
    cvt_f32_bf16<<<(n + 255) / 256, 256, 0, stream>>>(W_o, Wob, n);
  }

  // 2) kqv projection: (4096x512) x (512x2560), 32x32 tiles per wave
  {
    int tiles = (MROWS / 32) * (NKQV / 32);
    gemm_bf16_wmma<<<(tiles + 7) / 8, 256, 0, stream>>>(Xb, Wkqvb, b_kqv, kqv,
                                                        MROWS, NKQV, KDIM);
  }
  // 3) p projection: (4096x512) x (512x96)
  {
    int tiles = (MROWS / 32) * (NP / 32);
    gemm_bf16_wmma<<<(tiles + 7) / 8, 256, 0, stream>>>(Xb, Wpb, b_p, p,
                                                        MROWS, NP, KDIM);
  }
  // 4) featurization
  featurize<<<MROWS * HH, 256, 0, stream>>>(kqv, p, term, kg, qy, dg, vb, db);

  // 5) recurrence + readout (+ final states/tick into d_out)
  recurrence<<<BB * HH, 256, 0, stream>>>(kg, qy, dg, vb, db,
                                          k_prev, v_prev, s_prev, tick,
                                          attnb, out);

  // 6) output projection: (4096x512) x (512x512) -> d_out[0 : T*B*IN]
  {
    int tiles = (MROWS / 32) * (IN_ / 32);
    gemm_bf16_wmma<<<(tiles + 7) / 8, 256, 0, stream>>>(attnb, Wob, b_o, out,
                                                        MROWS, IN_, KDIM);
  }
}
